// _AGRTMLPSwitchHardLite_14035953123953
// MI455X (gfx1250) — matrix operand order repaired and checked by reading, not compiled, not run on hardware
//
#include <hip/hip_runtime.h>

typedef __bf16 bf16;
typedef __attribute__((ext_vector_type(16))) __bf16 v16bf;
typedef __attribute__((ext_vector_type(8)))  __bf16 v8bf;
typedef __attribute__((ext_vector_type(8)))  float  v8f;

#define LDA   576     // padded A stride (supports din up to 544)
#define EPSC  1e-5f
#define ROWS  64      // rows per workgroup
#define SUBT  (ROWS/16)

union ABFrag { v16bf v; v8bf h[2]; };

// ---- WMMA: SUBT 16x16 C tiles (row subtiles) sharing each B fragment --------------
// K loop deliberately capped at unroll 2: A fragments are re-read from LDS each
// k-block (ds_load_b128) instead of being fully preloaded (which spills at SUBT=4).
template<int S>
__device__ __forceinline__ void wmma_tileN(const bf16* sAp, int lda,
                                           const bf16* __restrict__ Wp, int doutP,
                                           int ntile, int kblocks, int lane, v8f (&c)[S])
{
  const int hf   = (lane >> 4) & 1;
  const int rowA = lane & 15;          // A: M = lane&15
  const int kAdd = hf ? 8 : 0;         // A: lanes>=16 carry K {8..15, 24..31}
  const int ncol = ntile*16 + (lane & 15);
  const int bOff = hf ? 8 : 0;         // B: lanes>=16 carry K {8..15, 24..31}
#pragma unroll 2
  for (int kb = 0; kb < kblocks; ++kb) {
    ABFrag b;
    const bf16* bp = Wp + ((size_t)kb*doutP + ncol)*32 + bOff;
    b.h[0] = *(const v8bf*)(bp);
    b.h[1] = *(const v8bf*)(bp + 16);
    if (kb + 1 < kblocks)
      __builtin_prefetch(Wp + ((size_t)(kb+1)*doutP + ncol)*32, 0, 0);
#pragma unroll
    for (int s = 0; s < S; ++s) {
      ABFrag a;
      const bf16* ap = sAp + (rowA + 16*s)*lda + kb*32 + kAdd;
      a.h[0] = *(const v8bf*)(ap);
      a.h[1] = *(const v8bf*)(ap + 16);
      c[s] = __builtin_amdgcn_wmma_f32_16x16x32_bf16(false, a.v, false, b.v,
                                                     (short)0, c[s], false, false);
    }
  }
}

// epilogue: bias + (relu | sigmoid) -> bf16 LDS for all row subtiles
template<int S>
__device__ __forceinline__ void store_tileN(v8f (&c)[S], bf16* sOut, int ldo,
                                            const float* __restrict__ bias, int ntile,
                                            int doutTrue, int lane, int act)
{
  const int   ncol  = ntile*16 + (lane & 15);
  const float bv    = (ncol < doutTrue) ? bias[ncol] : 0.f;
  const int   rbase = (lane < 16) ? 0 : 8;
#pragma unroll
  for (int s = 0; s < S; ++s) {
#pragma unroll
    for (int i = 0; i < 8; ++i) {
      float v = c[s][i] + bv;
      v = act ? (1.f/(1.f + __expf(-v))) : fmaxf(v, 0.f);
      sOut[(16*s + rbase + i)*ldo + ncol] = (bf16)v;
    }
  }
}

#define ZEROC(c) { _Pragma("unroll") for (int s_ = 0; s_ < SUBT; ++s_) \
                   c[s_] = (v8f){0.f,0.f,0.f,0.f,0.f,0.f,0.f,0.f}; }

// half-wave (16-lane) reductions
__device__ __forceinline__ float hadd(float v){
  v += __shfl_xor(v,1,32); v += __shfl_xor(v,2,32);
  v += __shfl_xor(v,4,32); v += __shfl_xor(v,8,32);
  return v;
}
__device__ __forceinline__ float hmax(float v){
  v = fmaxf(v,__shfl_xor(v,1,32)); v = fmaxf(v,__shfl_xor(v,2,32));
  v = fmaxf(v,__shfl_xor(v,4,32)); v = fmaxf(v,__shfl_xor(v,8,32));
  return v;
}

// rf slots: 0 gm,1 gmx,2 gsd, 3 tkm,4 tkmx,5 tksd,6 tkc, 7 dcross, 8 ybase,
// 9 hsum,10 hsumsq, 11 dsafe, 12 e1,13 e2, 14 amax,15 aent,16 egap,17 dexp,
// 18 beta(raw->sig), 20 mu,21 rs (current LN stage)
__device__ const int EMAP_RT[4]  = {8,0,1,2};
__device__ const int EMAP_MG[11] = {8,0,1,2,14,15,16,3,4,5,6};

__device__ __forceinline__ void build_ln512(bf16* sA, const bf16* sH,
                                            const float* __restrict__ g,
                                            const float* __restrict__ b,
                                            float (*rf)[24], int t)
{
  for (int i = t; i < ROWS*512; i += 256) {
    int r = i >> 9, c = i & 511;
    float v = ((float)sH[r*512+c] - rf[r][20]) * rf[r][21] * g[c] + b[c];
    sA[r*LDA + c] = (bf16)v;
  }
}

__device__ __forceinline__ void build_ln544(bf16* sA, const bf16* sH,
                                            const float* __restrict__ g,
                                            const float* __restrict__ b,
                                            float (*rf)[24],
                                            const int* emap, int dinTrue, int t)
{
  for (int i = t; i < ROWS*544; i += 256) {
    int r = i / 544, c = i - r*544;
    float val = 0.f;
    if (c < dinTrue) {
      float v = (c < 512) ? (float)sH[r*512+c] : rf[r][emap[c-512]];
      val = (v - rf[r][20]) * rf[r][21] * g[c] + b[c];
    }
    sA[r*LDA + c] = (bf16)val;
  }
}

// per-row dot hidden[hidTrue] . w2 (+b2) -> rf[r][slot]
__device__ __forceinline__ void rowdot1(const bf16* sHid,
                                        const float* __restrict__ w2, const float* __restrict__ b2,
                                        int hidTrue, float (*rf)[24], int slot,
                                        int wave, int lane)
{
  int hf = lane >> 4;
#pragma unroll
  for (int rr = 0; rr < ROWS/16; ++rr) {
    int r = 16*rr + wave*2 + hf;
    float acc = 0.f;
    for (int h = lane & 15; h < hidTrue; h += 16)
      acc += (float)sHid[r*512 + h] * w2[h];
    acc = hadd(acc);
    if ((lane & 15) == 0) rf[r][slot] = acc + b2[0];
  }
}

// ---------------- weight bf16 pack into WMMA-B-fragment layout ---------------------
__global__ void pack_w_bf16(const float* __restrict__ W, bf16* __restrict__ dst,
                            int din, int dout, int dinP, int doutP)
{
  int idx = blockIdx.x * 256 + threadIdx.x;
  int total = (dinP >> 5) * doutP * 32;
  if (idx >= total) return;
  int kk   = idx & 31;
  int rest = idx >> 5;
  int n    = rest % doutP;
  int kb   = rest / doutP;
  int k    = kb*32 + kk;
  float v  = (k < din && n < dout) ? W[(size_t)k*dout + n] : 0.f;
  dst[idx] = (bf16)v;
}

// ---------------- all parameters ---------------------------------------------------
struct P {
  const float* x;
  const bf16 *ig1, *ig2, *bb, *sw1, *e1w, *e2w, *rtw, *btw, *mgw;
  const float *ig_b1, *ig_b2, *bb_b, *base_w, *base_b;
  const float *safe_g, *safe_be, *safe_b1, *safe_w2, *safe_b2;
  const float *e1_g, *e1_be, *e1_b1, *e1_w2, *e1_b2;
  const float *e2_g, *e2_be, *e2_b1, *e2_w2, *e2_b2;
  const float *rt_g, *rt_be, *rt_b1, *rt_w2, *rt_b2;
  const float *bt_g, *bt_be, *bt_b1, *bt_w2, *bt_b2;
  const float *mg_g, *mg_be, *mg_b1, *mg_w2, *mg_b2;
  const float *ca_sp_w, *ca_sp_b, *ca_cp_w, *ca_cp_b;
  const float *ca_g, *ca_be, *ca_w1, *ca_b1, *ca_w2, *ca_b2;
  const float *es, *cs;
  float* out;
};

// ---------------- fused forward: 64 rows / workgroup -------------------------------
__global__ __launch_bounds__(256) void fused_fwd(P p)
{
  __shared__ __align__(16) bf16 sA[ROWS*LDA];   // x / x*g / LN'd inputs
  __shared__ __align__(16) bf16 sH[ROWS*512];   // h1 then h_base
  __shared__ __align__(16) bf16 sG[ROWS*512];   // g then hidden activations
  __shared__ float rf[ROWS][24];

  const int t    = threadIdx.x;
  const int lane = t & 31, wave = t >> 5;
  const int row0 = blockIdx.x * ROWS;

  // Phase 0: load x tile -> bf16 LDS
  for (int i = t; i < ROWS*512; i += 256) {
    int r = i >> 9, c = i & 511;
    sA[r*LDA + c] = (bf16)p.x[(size_t)(row0 + r)*512 + c];
  }
  __syncthreads();

  // Phase 1: h1 = relu(x @ ig_w1 + b1)   [ROWSx256]
  for (int nt = wave; nt < 16; nt += 8) {
    v8f c[SUBT]; ZEROC(c);
    wmma_tileN(sA, LDA, p.ig1, 256, nt, 16, lane, c);
    store_tileN(c, sH, 512, p.ig_b1, nt, 256, lane, 0);
  }
  __syncthreads();

  // Phase 2: g = sigmoid(h1 @ ig_w2 + b2) [ROWSx512]
  for (int nt = wave; nt < 32; nt += 8) {
    v8f c[SUBT]; ZEROC(c);
    wmma_tileN(sH, 512, p.ig2, 512, nt, 8, lane, c);
    store_tileN(c, sG, 512, p.ig_b2, nt, 512, lane, 1);
  }
  __syncthreads();

  // Phase 2b: gate stats (mean/max/std) per row
  {
    int hf = lane >> 4;
#pragma unroll
    for (int rr = 0; rr < ROWS/16; ++rr) {
      int r = 16*rr + wave*2 + hf;
      float s = 0.f, ss = 0.f, mx = -1e30f;
      for (int c = lane & 15; c < 512; c += 16) {
        float g = (float)sG[r*512 + c];
        s += g; ss += g*g; mx = fmaxf(mx, g);
      }
      s = hadd(s); ss = hadd(ss); mx = hmax(mx);
      if ((lane & 15) == 0) {
        float mean = s * (1.f/512.f);
        rf[r][0] = mean; rf[r][1] = mx;
        rf[r][2] = sqrtf(fmaxf(ss*(1.f/512.f) - mean*mean, 0.f));
      }
    }
  }
  __syncthreads();

  // Phase 3: x_gated = x*g (in place), top-8 + cross adapter (one row per lane)
  for (int i = t; i < ROWS*512; i += 256) {
    int r = i >> 9, c = i & 511;
    sA[r*LDA + c] = (bf16)((float)sA[r*LDA + c] * (float)sG[r*512 + c]);
  }
  if (wave < 2) {
    int r = wave*32 + lane;
    float tg[8]; int ti[8];
#pragma unroll
    for (int j = 0; j < 8; ++j) { tg[j] = -1e30f; ti[j] = 0; }
    for (int c = 0; c < 512; ++c) {
      float g = (float)sG[r*512 + c];
      if (g > tg[7]) {
        int j = 7;
        while (j > 0 && tg[j-1] < g) { tg[j] = tg[j-1]; ti[j] = ti[j-1]; --j; }
        tg[j] = g; ti[j] = c;
      }
    }
    float tsum = 0.f, tmx = -1e30f, tss = 0.f;
#pragma unroll
    for (int j = 0; j < 8; ++j) { tsum += tg[j]; tss += tg[j]*tg[j]; tmx = fmaxf(tmx, tg[j]); }
    float tmean = tsum*(1.f/8.f);
    float tstd  = sqrtf(fmaxf(tss*(1.f/8.f) - tmean*tmean, 0.f));
    rf[r][3] = tmean; rf[r][4] = tmx; rf[r][5] = tstd; rf[r][6] = tmx/(tmean + 1e-6f);

    float xw[8], inv = 1.f/(tsum + 1e-6f);
#pragma unroll
    for (int j = 0; j < 8; ++j)
      xw[j] = p.x[(size_t)(row0 + r)*512 + ti[j]] * tg[j]*inv;
    float xwsum = xw[0]+xw[1]+xw[2]+xw[3]+xw[4]+xw[5]+xw[6]+xw[7];
    float pooled[8];
#pragma unroll
    for (int q = 0; q < 8; ++q)
      pooled[q] = xwsum*(1.f/8.f)*p.ca_sp_w[q] + p.ca_sp_b[q];
    float z[8];
#pragma unroll
    for (int q = 0; q < 8; ++q) {
      float d = p.ca_cp_b[q];
#pragma unroll
      for (int c2 = 0; c2 < 8; ++c2) d += pooled[c2]*p.ca_cp_w[c2*8 + q];
      z[q] = pooled[q] + pooled[q]*d;
    }
    float mu = 0.f;
#pragma unroll
    for (int q = 0; q < 8; ++q) mu += z[q];
    mu *= (1.f/8.f);
    float var = 0.f;
#pragma unroll
    for (int q = 0; q < 8; ++q) { float d = z[q]-mu; var += d*d; }
    float rs = rsqrtf(var*(1.f/8.f) + EPSC);
    float lnz[8];
#pragma unroll
    for (int q = 0; q < 8; ++q) lnz[q] = (z[q]-mu)*rs*p.ca_g[q] + p.ca_be[q];
    float outv = p.ca_b2[0];
    for (int hh = 0; hh < 16; ++hh) {
      float a = p.ca_b1[hh];
#pragma unroll
      for (int q = 0; q < 8; ++q) a += lnz[q]*p.ca_w1[q*16 + hh];
      outv += fmaxf(a, 0.f)*p.ca_w2[hh];
    }
    rf[r][7] = p.cs[0]*tanhf(outv);
  }
  __syncthreads();

  // Phase 4: h_base = relu(x_gated @ bb_w + bb_b) [ROWSx512]
  for (int nt = wave; nt < 32; nt += 8) {
    v8f c[SUBT]; ZEROC(c);
    wmma_tileN(sA, LDA, p.bb, 512, nt, 16, lane, c);
    store_tileN(c, sH, 512, p.bb_b, nt, 512, lane, 0);
  }
  __syncthreads();

  // Phase 4b: h stats + y_base
  {
    int hf = lane >> 4;
#pragma unroll
    for (int rr = 0; rr < ROWS/16; ++rr) {
      int r = 16*rr + wave*2 + hf;
      float s = 0.f, ss = 0.f, yb = 0.f;
      for (int c = lane & 15; c < 512; c += 16) {
        float h = (float)sH[r*512 + c];
        s += h; ss += h*h; yb += h*p.base_w[c];
      }
      s = hadd(s); ss = hadd(ss); yb = hadd(yb);
      if ((lane & 15) == 0) {
        rf[r][9] = s; rf[r][10] = ss; rf[r][8] = yb + p.base_b[0];
        float mu = s*(1.f/512.f);
        rf[r][20] = mu;
        rf[r][21] = rsqrtf(ss*(1.f/512.f) - mu*mu + EPSC);
      }
    }
  }
  __syncthreads();

  // Phase 5: delta_safe
  build_ln512(sA, sH, p.safe_g, p.safe_be, rf, t);
  __syncthreads();
  for (int nt = wave; nt < 16; nt += 8) {
    v8f c[SUBT]; ZEROC(c);
    wmma_tileN(sA, LDA, p.sw1, 256, nt, 16, lane, c);
    store_tileN(c, sG, 512, p.safe_b1, nt, 256, lane, 0);
  }
  __syncthreads();
  rowdot1(sG, p.safe_w2, p.safe_b2, 256, rf, 11, wave, lane);
  __syncthreads();

  // Phase 6/7: experts e1, e2
  build_ln512(sA, sH, p.e1_g, p.e1_be, rf, t);
  __syncthreads();
  for (int nt = wave; nt < 8; nt += 8) {
    v8f c[SUBT]; ZEROC(c);
    wmma_tileN(sA, LDA, p.e1w, 128, nt, 16, lane, c);
    store_tileN(c, sG, 512, p.e1_b1, nt, 128, lane, 0);
  }
  __syncthreads();
  rowdot1(sG, p.e1_w2, p.e1_b2, 128, rf, 12, wave, lane);
  __syncthreads();

  build_ln512(sA, sH, p.e2_g, p.e2_be, rf, t);
  __syncthreads();
  for (int nt = wave; nt < 8; nt += 8) {
    v8f c[SUBT]; ZEROC(c);
    wmma_tileN(sA, LDA, p.e2w, 128, nt, 16, lane, c);
    store_tileN(c, sG, 512, p.e2_b1, nt, 128, lane, 0);
  }
  __syncthreads();
  rowdot1(sG, p.e2_w2, p.e2_b2, 128, rf, 13, wave, lane);
  __syncthreads();

  // Phase 8: router (din 516 -> pad 544, hid 129 -> pad 144, 2 logits)
  if (wave < 2) {
    int r = wave*32 + lane;
    float e0 = rf[r][8], e1v = rf[r][0], e2v = rf[r][1], e3 = rf[r][2];
    float s  = rf[r][9]  + e0 + e1v + e2v + e3;
    float ss = rf[r][10] + e0*e0 + e1v*e1v + e2v*e2v + e3*e3;
    float mu = s*(1.f/516.f);
    rf[r][20] = mu;
    rf[r][21] = rsqrtf(ss*(1.f/516.f) - mu*mu + EPSC);
  }
  __syncthreads();
  build_ln544(sA, sH, p.rt_g, p.rt_be, rf, EMAP_RT, 516, t);
  __syncthreads();
  for (int nt = wave; nt < 9; nt += 8) {
    v8f c[SUBT]; ZEROC(c);
    wmma_tileN(sA, LDA, p.rtw, 144, nt, 17, lane, c);
    store_tileN(c, sG, 512, p.rt_b1, nt, 129, lane, 0);
  }
  __syncthreads();
  {
    int hf = lane >> 4;
#pragma unroll
    for (int rr = 0; rr < ROWS/16; ++rr) {
      int r = 16*rr + wave*2 + hf;
      float a0 = 0.f, a1 = 0.f;
      for (int h = lane & 15; h < 129; h += 16) {
        float hv = (float)sG[r*512 + h];
        a0 += hv*p.rt_w2[h*2];
        a1 += hv*p.rt_w2[h*2 + 1];
      }
      a0 = hadd(a0); a1 = hadd(a1);
      if ((lane & 15) == 0) {
        float l0 = (a0 + p.rt_b2[0])*(1.f/1.5f);
        float l1 = (a1 + p.rt_b2[1])*(1.f/1.5f);
        float m  = fmaxf(l0, l1);
        float x0 = __expf(l0 - m), x1 = __expf(l1 - m);
        float inv = 1.f/(x0 + x1);
        float al0 = x0*inv, al1 = x1*inv;
        rf[r][14] = fmaxf(al0, al1);
        rf[r][15] = -(al0*__logf(fmaxf(al0, 1e-8f)) + al1*__logf(fmaxf(al1, 1e-8f)));
        float ev1 = rf[r][12], ev2 = rf[r][13];
        rf[r][16] = fabsf(ev1 - ev2);
        rf[r][17] = p.es[0]*tanhf(al0*ev1 + al1*ev2);
      }
    }
  }
  __syncthreads();

  // Phase 9: beta head (same input stats as router)
  build_ln544(sA, sH, p.bt_g, p.bt_be, rf, EMAP_RT, 516, t);
  __syncthreads();
  for (int nt = wave; nt < 9; nt += 8) {
    v8f c[SUBT]; ZEROC(c);
    wmma_tileN(sA, LDA, p.btw, 144, nt, 17, lane, c);
    store_tileN(c, sG, 512, p.bt_b1, nt, 129, lane, 0);
  }
  __syncthreads();
  rowdot1(sG, p.bt_w2, p.bt_b2, 129, rf, 18, wave, lane);
  __syncthreads();

  // Phase 10: mode gate (din 523 -> 544, hid 130 -> 144, 4 logits)
  if (wave < 2) {
    int r = wave*32 + lane;
    rf[r][18] = 1.f/(1.f + __expf(-rf[r][18]));   // beta
    float s = rf[r][9], ss = rf[r][10];
#pragma unroll
    for (int j = 0; j < 11; ++j) {
      float v = rf[r][EMAP_MG[j]];
      s += v; ss += v*v;
    }
    float mu = s*(1.f/523.f);
    rf[r][20] = mu;
    rf[r][21] = rsqrtf(ss*(1.f/523.f) - mu*mu + EPSC);
  }
  __syncthreads();
  build_ln544(sA, sH, p.mg_g, p.mg_be, rf, EMAP_MG, 523, t);
  __syncthreads();
  for (int nt = wave; nt < 9; nt += 8) {
    v8f c[SUBT]; ZEROC(c);
    wmma_tileN(sA, LDA, p.mgw, 144, nt, 17, lane, c);
    store_tileN(c, sG, 512, p.mg_b1, nt, 130, lane, 0);
  }
  __syncthreads();
  {
    int hf = lane >> 4;
#pragma unroll
    for (int rr = 0; rr < ROWS/16; ++rr) {
      int r = 16*rr + wave*2 + hf;
      float a0 = 0.f, a1 = 0.f, a2 = 0.f, a3 = 0.f;
      for (int h = lane & 15; h < 130; h += 16) {
        float hv = (float)sG[r*512 + h];
        a0 += hv*p.mg_w2[h*4];     a1 += hv*p.mg_w2[h*4 + 1];
        a2 += hv*p.mg_w2[h*4 + 2]; a3 += hv*p.mg_w2[h*4 + 3];
      }
      a0 = hadd(a0); a1 = hadd(a1); a2 = hadd(a2); a3 = hadd(a3);
      if ((lane & 15) == 0) {
        float l0 = (a0 + p.mg_b2[0])*(1.f/0.35f);
        float l1 = (a1 + p.mg_b2[1])*(1.f/0.35f);
        float l2 = (a2 + p.mg_b2[2])*(1.f/0.35f);
        float l3 = (a3 + p.mg_b2[3])*(1.f/0.35f);
        float m = fmaxf(fmaxf(l0, l1), fmaxf(l2, l3));
        float x0 = __expf(l0-m), x1 = __expf(l1-m), x2 = __expf(l2-m), x3 = __expf(l3-m);
        float inv = 1.f/(x0 + x1 + x2 + x3);
        float m1 = x1*inv, m2 = x2*inv, m3 = x3*inv;
        float dmix = tanhf(m1*rf[r][11] + m2*rf[r][17] + m3*rf[r][7]);
        p.out[row0 + r] = rf[r][8] + rf[r][18]*dmix;
      }
    }
  }
}

// ---------------- host side --------------------------------------------------------
enum : int {
  OFF_IG1 = 0,
  OFF_IG2 = OFF_IG1 + 512*256,
  OFF_BB  = OFF_IG2 + 256*512,
  OFF_SW1 = OFF_BB  + 512*512,
  OFF_E1  = OFF_SW1 + 512*256,
  OFF_E2  = OFF_E1  + 512*128,
  OFF_RT  = OFF_E2  + 512*128,
  OFF_BT  = OFF_RT  + 544*144,
  OFF_MG  = OFF_BT  + 544*144,
  WS_ELEMS = OFF_MG + 544*144
};

extern "C" void kernel_launch(void* const* d_in, const int* in_sizes, int n_in,
                              void* d_out, int out_size, void* d_ws, size_t ws_size,
                              hipStream_t stream)
{
  (void)in_sizes; (void)n_in; (void)out_size; (void)ws_size;
  bf16* ws = (bf16*)d_ws;
  auto f = [&](int i) { return (const float*)d_in[i]; };

  auto pack = [&](int idx, int off, int din, int dout, int dinP, int doutP) {
    int total = (dinP >> 5)*doutP*32;
    pack_w_bf16<<<(total + 255)/256, 256, 0, stream>>>(f(idx), ws + off, din, dout, dinP, doutP);
  };
  pack(1,  OFF_IG1, 512, 256, 512, 256);
  pack(3,  OFF_IG2, 256, 512, 256, 512);
  pack(5,  OFF_BB,  512, 512, 512, 512);
  pack(11, OFF_SW1, 512, 256, 512, 256);
  pack(17, OFF_E1,  512, 128, 512, 128);
  pack(23, OFF_E2,  512, 128, 512, 128);
  pack(29, OFF_RT,  516, 129, 544, 144);
  pack(35, OFF_BT,  516, 129, 544, 144);
  pack(41, OFF_MG,  523, 130, 544, 144);

  P p;
  p.x = f(0);
  p.ig1 = ws + OFF_IG1; p.ig2 = ws + OFF_IG2; p.bb = ws + OFF_BB;
  p.sw1 = ws + OFF_SW1; p.e1w = ws + OFF_E1;  p.e2w = ws + OFF_E2;
  p.rtw = ws + OFF_RT;  p.btw = ws + OFF_BT;  p.mgw = ws + OFF_MG;
  p.ig_b1 = f(2);  p.ig_b2 = f(4);  p.bb_b = f(6);  p.base_w = f(7); p.base_b = f(8);
  p.safe_g = f(9);  p.safe_be = f(10); p.safe_b1 = f(12); p.safe_w2 = f(13); p.safe_b2 = f(14);
  p.e1_g = f(15); p.e1_be = f(16); p.e1_b1 = f(18); p.e1_w2 = f(19); p.e1_b2 = f(20);
  p.e2_g = f(21); p.e2_be = f(22); p.e2_b1 = f(24); p.e2_w2 = f(25); p.e2_b2 = f(26);
  p.rt_g = f(27); p.rt_be = f(28); p.rt_b1 = f(30); p.rt_w2 = f(31); p.rt_b2 = f(32);
  p.bt_g = f(33); p.bt_be = f(34); p.bt_b1 = f(36); p.bt_w2 = f(37); p.bt_b2 = f(38);
  p.mg_g = f(39); p.mg_be = f(40); p.mg_b1 = f(42); p.mg_w2 = f(43); p.mg_b2 = f(44);
  p.ca_sp_w = f(45); p.ca_sp_b = f(46); p.ca_cp_w = f(47); p.ca_cp_b = f(48);
  p.ca_g = f(49); p.ca_be = f(50); p.ca_w1 = f(51); p.ca_b1 = f(52);
  p.ca_w2 = f(53); p.ca_b2 = f(54);
  p.es = f(55); p.cs = f(56);
  p.out = (float*)d_out;

  fused_fwd<<<65536/ROWS, 256, 0, stream>>>(p);
}
